// MambaNet_51891794871151
// MI455X (gfx1250) — compile-verified, matrix operand order checked by reading, not run on hardware
//
#include <hip/hip_runtime.h>

// ---------------- model dims ----------------
#define VOCAB 6
#define C_IN  4
#define DMODEL 256
#define NLAYER 4
#define TLEN  4096
#define BSZ   2
#define DI    512            // expand*D
#define DS    16             // d_state
#define DCONV 4
#define DTR   16             // dt_rank
#define MROWS (BSZ*TLEN)     // 8192

typedef __attribute__((ext_vector_type(16))) _Float16 v16h;
typedef __attribute__((ext_vector_type(8)))  _Float16 h8;
typedef __attribute__((ext_vector_type(8)))  float    v8f;

// Assemble a WMMA 16-half fragment from two contiguous 16B chunks.
__device__ __forceinline__ v16h load_frag(const _Float16* p) {
    h8 lo = *(const h8*)(p);
    h8 hi = *(const h8*)(p + 16);
    return __builtin_shufflevector(lo, hi,
        0,1,2,3,4,5,6,7,8,9,10,11,12,13,14,15);
}

// =====================================================================
// WMMA GEMM:  C[M,N] = A[M,K]*W[N,K]^T (+bias)(+softplus).
// A, W pre-converted to f16 and zero-padded so there are NO guards in
// the K-loop: per lane a fragment is 2x global_load_b128.
// Each wave computes NT adjacent 16x16 N-tiles sharing one A fragment;
// all NT B fragments are loaded before the wmma chain so the scheduler
// can stagger s_wait_loadcnt instead of draining before every wmma.
// Per ISA 7.12.2 16-bit A layout: lane<16 -> K {0-7,16-23},
// lane>=16 -> K {8-15,24-31}; B fragment mirrored over N.
// C/D layout: VGPR r -> M = r + 8*(lane>=16), N = lane&15.
// =====================================================================
template<int NT, int EPI>   // EPI: 0=bias, 1=bias+softplus
__global__ void gemm_wmma(const _Float16* __restrict__ Ah,
                          const _Float16* __restrict__ Wh,
                          const float* __restrict__ bias,
                          float* __restrict__ C,
                          _Float16* __restrict__ Ch,   // optional f16 copy
                          int M, int Neff, int nGroupsN, int K,
                          int lda, int ldw, int ldc, int ldch, int Nh)
{
    const int gwave = (blockIdx.x * blockDim.x + threadIdx.x) >> 5;
    const int lane  = threadIdx.x & 31;
    const int nTilesM = M >> 4;
    if (gwave >= nTilesM * nGroupsN) return;
    const int tm = gwave / nGroupsN;
    const int tg = gwave % nGroupsN;

    const int khalf = lane >> 4;
    const int mr    = lane & 15;
    const _Float16* ap = Ah + (size_t)(tm * 16 + mr) * lda + khalf * 8;
    const _Float16* wp[NT];
#pragma unroll
    for (int j = 0; j < NT; ++j)
        wp[j] = Wh + (size_t)((tg * NT + j) * 16 + mr) * ldw + khalf * 8;

    v8f acc[NT];
#pragma unroll
    for (int j = 0; j < NT; ++j) acc[j] = (v8f){};

    for (int kb = 0; kb < K; kb += 32) {
        // issue all loads for this K-step first ...
        v16h a = load_frag(ap + kb);
        v16h b[NT];
#pragma unroll
        for (int j = 0; j < NT; ++j)
            b[j] = load_frag(wp[j] + kb);
        __builtin_prefetch(ap + kb + 64, 0, 1);        // global_prefetch_b8
        // ... then the back-to-back wmma chain
#pragma unroll
        for (int j = 0; j < NT; ++j)
            acc[j] = __builtin_amdgcn_wmma_f32_16x16x32_f16(
                false, a, false, b[j], (short)0, acc[j], false, false);
    }

    const int mofs = khalf * 8;
#pragma unroll
    for (int j = 0; j < NT; ++j) {
        const int n = (tg * NT + j) * 16 + mr;
        if (n >= Neff) continue;                       // ragged heads only
        const float bv = bias ? bias[n] : 0.f;
#pragma unroll
        for (int r = 0; r < 8; ++r) {
            const int m = tm * 16 + mofs + r;
            float v = acc[j][r] + bv;
            if (EPI == 1) v = (v > 20.f) ? v : log1pf(__expf(v)); // softplus
            C[(size_t)m * ldc + n] = v;
            if (Ch && n < Nh) Ch[(size_t)m * ldch + n] = (_Float16)v;
        }
    }
}

// =====================================================================
// f32 -> f16 conversion with row/col zero-padding (weights staging).
// out is (Rpad, Kpad); in is (Reff, Keff) row-major.
// =====================================================================
__global__ void cvt_pad(const float* __restrict__ in, _Float16* __restrict__ out,
                        int Reff, int Rpad, int Keff, int Kpad)
{
    const int tid = blockIdx.x * blockDim.x + threadIdx.x;
    if (tid >= Rpad * Kpad) return;
    const int r = tid / Kpad, k = tid - r * Kpad;
    out[tid] = (r < Reff && k < Keff) ? (_Float16)in[(size_t)r * Keff + k]
                                      : (_Float16)0.f;
}

// =====================================================================
// x = emb[seq] + input @ inproj_w^T + inproj_b   (f32 + f16 copies)
// =====================================================================
__global__ void embed_kernel(const int* __restrict__ seq,
                             const float* __restrict__ inp,
                             const float* __restrict__ emb,
                             const float* __restrict__ w,
                             const float* __restrict__ b,
                             float* __restrict__ x, _Float16* __restrict__ xh)
{
    const int tid = blockIdx.x * blockDim.x + threadIdx.x;
    if (tid >= MROWS * DMODEL) return;
    const int d  = tid & (DMODEL - 1);
    const int bt = tid / DMODEL;
    float acc = emb[seq[bt] * DMODEL + d] + b[d];
#pragma unroll
    for (int c = 0; c < C_IN; ++c)
        acc += inp[bt * C_IN + c] * w[d * C_IN + c];
    x[tid]  = acc;
    xh[tid] = (_Float16)acc;
}

// =====================================================================
// depthwise causal conv1d + SiLU; writes f32 (scan) and f16 (x_proj A)
// =====================================================================
__global__ void conv_kernel(const float* __restrict__ xz,
                            const float* __restrict__ cw,
                            const float* __restrict__ cb,
                            float* __restrict__ xc, _Float16* __restrict__ xch)
{
    const int tid = blockIdx.x * blockDim.x + threadIdx.x;
    if (tid >= MROWS * DI) return;
    const int d  = tid & (DI - 1);
    const int bt = tid / DI;
    const int t  = bt & (TLEN - 1);
    const int bb = bt / TLEN;
    float acc = cb[d];
#pragma unroll
    for (int j = 0; j < DCONV; ++j) {
        const int tt = t - (DCONV - 1) + j;
        if (tt >= 0)
            acc += cw[d * DCONV + j] * xz[((size_t)(bb * TLEN + tt)) * (2 * DI) + d];
    }
    acc = acc / (1.f + __expf(-acc));   // silu
    xc[tid]  = acc;
    xch[tid] = (_Float16)acc;
}

// =====================================================================
// selective scan: lane = (channel pair, state); shfl reduce over 16
// states; fused D-skip + silu(z) gate; writes f16 y for out_proj GEMM.
// =====================================================================
__global__ void scan_kernel(const float* __restrict__ dt,
                            const float* __restrict__ xd,  // (B*T,48) [dt|B|C]
                            const float* __restrict__ xc,
                            const float* __restrict__ xz,  // z at [.., DI:2DI]
                            const float* __restrict__ A_log,
                            const float* __restrict__ Dp,
                            _Float16* __restrict__ yh)
{
    const int tid  = blockIdx.x * blockDim.x + threadIdx.x;
    const int lane = tid & 31;
    const int s    = lane & 15;
    const int wave = tid >> 5;
    const int ch   = wave * 2 + (lane >> 4);
    if (ch >= BSZ * DI) return;
    const int bb = ch / DI;
    const int d  = ch & (DI - 1);

    const float Aval = -__expf(A_log[d * DS + s]);
    const float Dv   = Dp[d];
    float h = 0.f;

    for (int t = 0; t < TLEN; ++t) {
        const size_t bt = (size_t)bb * TLEN + t;
        const float dtv = dt[bt * DI + d];
        const float xcv = xc[bt * DI + d];
        const float Bv  = xd[bt * (DTR + 2 * DS) + DTR + s];
        const float Cv  = xd[bt * (DTR + 2 * DS) + DTR + DS + s];
        h = __expf(dtv * Aval) * h + dtv * Bv * xcv;
        float contrib = h * Cv;
#pragma unroll
        for (int off = 8; off >= 1; off >>= 1)
            contrib += __shfl_xor(contrib, off, 32);
        if (s == 0) {
            const float zv = xz[bt * (2 * DI) + DI + d];
            float yv = contrib + Dv * xcv;
            yv *= zv / (1.f + __expf(-zv));
            yh[bt * DI + d] = (_Float16)yv;
        }
    }
}

// =====================================================================
// x = layernorm(x + mx) * g + b ; also emit f16 copy for next GEMMs
// =====================================================================
__global__ void ln_kernel(float* __restrict__ x,
                          const float* __restrict__ mx,
                          const float* __restrict__ g,
                          const float* __restrict__ b,
                          _Float16* __restrict__ xh)
{
    __shared__ float red[DMODEL];
    const int row = blockIdx.x;
    const int tid = threadIdx.x;
    float v = x[(size_t)row * DMODEL + tid] + mx[(size_t)row * DMODEL + tid];

    red[tid] = v;
    __syncthreads();
    for (int off = DMODEL / 2; off > 0; off >>= 1) {
        if (tid < off) red[tid] += red[tid + off];
        __syncthreads();
    }
    const float mu = red[0] * (1.f / DMODEL);
    __syncthreads();

    const float c = v - mu;
    red[tid] = c * c;
    __syncthreads();
    for (int off = DMODEL / 2; off > 0; off >>= 1) {
        if (tid < off) red[tid] += red[tid + off];
        __syncthreads();
    }
    const float rstd = rsqrtf(red[0] * (1.f / DMODEL) + 1e-5f);
    const float o = c * rstd * g[tid] + b[tid];
    x[(size_t)row * DMODEL + tid]  = o;
    xh[(size_t)row * DMODEL + tid] = (_Float16)o;
}

// =====================================================================
extern "C" void kernel_launch(void* const* d_in, const int* in_sizes, int n_in,
                              void* d_out, int out_size, void* d_ws, size_t ws_size,
                              hipStream_t stream)
{
    const int*   seq      = (const int*)  d_in[0];
    const float* input    = (const float*)d_in[1];
    const float* emb      = (const float*)d_in[2];
    const float* inproj_w = (const float*)d_in[3];
    const float* inproj_b = (const float*)d_in[4];
    const float* blk_in_w = (const float*)d_in[5];   // (L,1024,256)
    const float* blk_in_b = (const float*)d_in[6];
    const float* conv_w   = (const float*)d_in[7];   // (L,512,4)
    const float* conv_b   = (const float*)d_in[8];
    const float* xproj_w  = (const float*)d_in[9];   // (L,48,512)
    const float* dt_w     = (const float*)d_in[10];  // (L,512,16)
    const float* dt_b     = (const float*)d_in[11];
    const float* A_log    = (const float*)d_in[12];  // (L,512,16)
    const float* Dp       = (const float*)d_in[13];
    const float* out_w    = (const float*)d_in[14];  // (L,256,512)
    const float* out_b    = (const float*)d_in[15];
    const float* ln_g     = (const float*)d_in[16];
    const float* ln_b     = (const float*)d_in[17];
    const float* head_w   = (const float*)d_in[18];  // (6,256)
    const float* head_b   = (const float*)d_in[19];
    const float* outp_w   = (const float*)d_in[20];  // (4,256)
    const float* outp_b   = (const float*)d_in[21];

    float* out = (float*)d_out;          // [head 8192*6 | outp 8192*4]

    // ---------------- workspace carve-up ----------------
    float* ws  = (float*)d_ws;
    float* x    = ws;                                      // 8192*256 f32
    float* xz   = x   + (size_t)MROWS * DMODEL;            // 8192*1024
    float* xc   = xz  + (size_t)MROWS * 2 * DI;            // 8192*512
    float* xd   = xc  + (size_t)MROWS * DI;                // 8192*48
    float* dtb  = xd  + (size_t)MROWS * (DTR + 2 * DS);    // 8192*512
    float* mx   = dtb + (size_t)MROWS * DI;                // 8192*256

    _Float16* hbase = (_Float16*)(mx + (size_t)MROWS * DMODEL);
    _Float16* xh     = hbase;                              // 8192*256
    _Float16* xch    = xh    + (size_t)MROWS * DMODEL;     // 8192*512
    _Float16* xdh    = xch   + (size_t)MROWS * DI;         // 8192*32 (K-padded)
    _Float16* yh     = xdh   + (size_t)MROWS * 32;         // 8192*512
    _Float16* inwh   = yh    + (size_t)MROWS * DI;         // 4*1024*256
    _Float16* xpwh   = inwh  + (size_t)NLAYER * 2 * DI * DMODEL;  // 4*48*512
    _Float16* dtwh   = xpwh  + (size_t)NLAYER * 48 * DI;   // 4*512*32 (K-padded)
    _Float16* outwh  = dtwh  + (size_t)NLAYER * DI * 32;   // 4*256*512
    _Float16* headwh = outwh + (size_t)NLAYER * DMODEL * DI; // 16*256 (N-padded)
    _Float16* outpwh = headwh + (size_t)16 * DMODEL;       // 16*256 (N-padded)

    auto blocks = [](long n) { return dim3((unsigned)((n + 255) / 256)); };
    auto ggrid  = [](int M, int nGroupsN) {
        const long waves = (long)(M >> 4) * nGroupsN;      // 1 wave / tile group
        return dim3((unsigned)((waves + 3) / 4));          // 4 waves / 128 thr
    };

    // zero K-pad region of xdh once per call (cols 16..31 never written)
    hipMemsetAsync(xdh, 0, (size_t)MROWS * 32 * sizeof(_Float16), stream);

    // ---- stage all weights to f16 (padded) ----
    cvt_pad<<<blocks(NLAYER*2*DI*DMODEL), 256, 0, stream>>>(blk_in_w, inwh,
        NLAYER*2*DI, NLAYER*2*DI, DMODEL, DMODEL);
    cvt_pad<<<blocks(NLAYER*48*DI), 256, 0, stream>>>(xproj_w, xpwh,
        NLAYER*48, NLAYER*48, DI, DI);
    cvt_pad<<<blocks(NLAYER*DI*32), 256, 0, stream>>>(dt_w, dtwh,
        NLAYER*DI, NLAYER*DI, DTR, 32);
    cvt_pad<<<blocks(NLAYER*DMODEL*DI), 256, 0, stream>>>(out_w, outwh,
        NLAYER*DMODEL, NLAYER*DMODEL, DI, DI);
    cvt_pad<<<blocks(16*DMODEL), 256, 0, stream>>>(head_w, headwh,
        VOCAB, 16, DMODEL, DMODEL);
    cvt_pad<<<blocks(16*DMODEL), 256, 0, stream>>>(outp_w, outpwh,
        C_IN, 16, DMODEL, DMODEL);

    // ---- embedding + input projection ----
    embed_kernel<<<blocks(MROWS*DMODEL), 256, 0, stream>>>(
        seq, input, emb, inproj_w, inproj_b, x, xh);

    for (int l = 0; l < NLAYER; ++l) {
        const float* ib = blk_in_b + (size_t)l * 2 * DI;
        const float* cw = conv_w   + (size_t)l * DI * DCONV;
        const float* cb = conv_b   + (size_t)l * DI;
        const float* db = dt_b     + (size_t)l * DI;
        const float* al = A_log    + (size_t)l * DI * DS;
        const float* dp = Dp       + (size_t)l * DI;
        const float* ob = out_b    + (size_t)l * DMODEL;

        // xz = x @ in_w^T + in_b   (M=8192,N=1024,K=256), NT=4
        gemm_wmma<4,0><<<ggrid(MROWS, 16), 128, 0, stream>>>(
            xh, inwh + (size_t)l*2*DI*DMODEL, ib, xz, nullptr,
            MROWS, 2*DI, 16, DMODEL, DMODEL, DMODEL, 2*DI, 0, 0);

        conv_kernel<<<blocks(MROWS*DI), 256, 0, stream>>>(xz, cw, cb, xc, xch);

        // x_dbl = xc @ xproj_w^T (N=48,K=512), NT=3; f16 copy of dt cols
        gemm_wmma<3,0><<<ggrid(MROWS, 1), 128, 0, stream>>>(
            xch, xpwh + (size_t)l*48*DI, nullptr, xd, xdh,
            MROWS, DTR+2*DS, 1, DI, DI, DI, DTR+2*DS, 32, DTR);

        // dt = softplus(x_dbl[:,:16] @ dt_w^T + dt_b)  (K padded 16->32)
        gemm_wmma<4,1><<<ggrid(MROWS, 8), 128, 0, stream>>>(
            xdh, dtwh + (size_t)l*DI*32, db, dtb, nullptr,
            MROWS, DI, 8, 32, 32, 32, DI, 0, 0);

        // selective scan + D-skip + silu(z) gate -> yh (f16)
        scan_kernel<<<blocks((long)BSZ*DI/2*32), 256, 0, stream>>>(
            dtb, xd, xc, xz, al, dp, yh);

        // mx = y @ out_w^T + out_b  (N=256,K=512), NT=4
        gemm_wmma<4,0><<<ggrid(MROWS, 4), 128, 0, stream>>>(
            yh, outwh + (size_t)l*DMODEL*DI, ob, mx, nullptr,
            MROWS, DMODEL, 4, DI, DI, DI, DMODEL, 0, 0);

        // x = layernorm(x + mx)
        ln_kernel<<<MROWS, DMODEL, 0, stream>>>(
            x, mx, ln_g + (size_t)l*DMODEL, ln_b + (size_t)l*DMODEL, xh);
    }

    // ---- heads (W zero-padded to 16 rows; store masked to Neff) ----
    gemm_wmma<1,0><<<ggrid(MROWS, 1), 128, 0, stream>>>(
        xh, headwh, head_b, out, nullptr,
        MROWS, VOCAB, 1, DMODEL, DMODEL, DMODEL, VOCAB, 0, 0);
    gemm_wmma<1,0><<<ggrid(MROWS, 1), 128, 0, stream>>>(
        xh, outpwh, outp_b, out + (size_t)MROWS * VOCAB, nullptr,
        MROWS, C_IN, 1, DMODEL, DMODEL, DMODEL, C_IN, 0, 0);
}
